// TauTrackFinderV2_12695923327033
// MI455X (gfx1250) — matrix operand order repaired and checked by reading, not compiled, not run on hardware
//
#include <hip/hip_runtime.h>
#include <math.h>

// ---------------------------------------------------------------------------
// Problem constants (from reference)
// ---------------------------------------------------------------------------
#define BB   8
#define PP   2048
#define CC   256
#define KK   16
#define MSGC 64
#define KCC  256
#define DD   128
#define HH   4
#define NLL  2
#define COMBP 352   // 322 padded to multiple of 32 for WMMA K
#define EPSV 1e-5f

typedef __attribute__((ext_vector_type(16))) _Float16 v16h;
typedef __attribute__((ext_vector_type(8)))  _Float16 v8h;
typedef __attribute__((ext_vector_type(8)))  float    v8f;

__device__ __forceinline__ float gelu_f(float x) {
    return 0.5f * x * (1.0f + erff(x * 0.70710678118654752f));
}

// ---------------------------------------------------------------------------
// Generic batched WMMA GEMM:  C[z][m][n] = scale * sum_k A[z][m][k]*B[z][n][k]
//                              (+bias[n]) (+=C if ACC) (gelu if GELU)
// A: row-major (M x Kp) f16, lda multiple of 32
// B: row-major (N x Kp) f16 (i.e. "B-transposed" storage), ldb multiple of 32
// C: f32, base offset per z = (z/zdiv)*sC1 + (z%zdiv)*sC2
// One wave computes one 16x16 tile via v_wmma_f32_16x16x32_f16.
// ---------------------------------------------------------------------------
#define GF_ACC  1
#define GF_GELU 2

__global__ void ttf_gemm(const _Float16* __restrict__ A, long long sA, int lda,
                         const _Float16* __restrict__ B, long long sB, int ldb,
                         float* __restrict__ C, long long sC1, long long sC2,
                         int zdiv, int ldc,
                         int M, int N, int Kp,
                         const float* __restrict__ bias, float scale, int flags)
{
    const int z = blockIdx.z;
    const _Float16* Ab = A + (long long)z * sA;
    const _Float16* Bb = B + (long long)z * sB;
    float* Cb = C + (long long)(z / zdiv) * sC1 + (long long)(z % zdiv) * sC2;

    const int wave = threadIdx.x >> 5;
    const int lane = threadIdx.x & 31;
    const int m0 = (blockIdx.y * 4 + wave) * 16;
    if (m0 >= M) return;
    const int n0 = blockIdx.x * 16;
    const int l15 = lane & 15;
    const int hi  = (lane & 16) ? 1 : 0;

    v8f acc = {};
    // A fragment: row m0+l15; K halves [base, base+8) and [base+16, base+24)
    const _Float16* pa = Ab + (size_t)(m0 + l15) * lda + (hi ? 8 : 0);
    // B fragment: row n0+l15; K halves [base, base+8) and [base+16, base+24)
    const _Float16* pb = Bb + (size_t)(n0 + l15) * ldb + (hi ? 8 : 0);

    for (int k0 = 0; k0 < Kp; k0 += 32) {
        v8h a0 = *(const v8h*)(pa);
        v8h a1 = *(const v8h*)(pa + 16);
        v8h b0 = *(const v8h*)(pb);
        v8h b1 = *(const v8h*)(pb + 16);
        v16h av, bv;
#pragma unroll
        for (int i = 0; i < 8; ++i) {
            av[i] = a0[i]; av[8 + i] = a1[i];
            bv[i] = b0[i]; bv[8 + i] = b1[i];
        }
        acc = __builtin_amdgcn_wmma_f32_16x16x32_f16(
            false, av, false, bv, (short)0, acc, false, false);
        pa += 32; pb += 32;
    }

    const int n = n0 + l15;
    const float bs = (bias != nullptr) ? bias[n] : 0.0f;
#pragma unroll
    for (int r = 0; r < 8; ++r) {
        const int m = m0 + r + hi * 8;
        float v = acc[r] * scale + bs;
        float* cp = Cb + (size_t)m * ldc + n;
        if (flags & GF_ACC)  v += *cp;
        if (flags & GF_GELU) v = gelu_f(v);
        *cp = v;
    }
}

// ---------------------------------------------------------------------------
// kNN: for each query point, top-16 smallest masked distances in (eta,phi)
// grid (B, P/256), block 256. Points tile staged in LDS.
// ---------------------------------------------------------------------------
__global__ void ttf_knn(const float* __restrict__ points,
                        const unsigned char* __restrict__ mask,
                        int* __restrict__ idx)
{
    __shared__ float sx[PP];
    __shared__ float sy[PP];
    __shared__ unsigned char sm[PP];
    const int b = blockIdx.x;
    const int tid = threadIdx.x;
    const float* px = points + (size_t)b * 2 * PP;
    for (int j = tid; j < PP; j += 256) {
        sx[j] = px[j];
        sy[j] = px[PP + j];
        sm[j] = mask[b * PP + j];
    }
    __syncthreads();
    const int p = blockIdx.y * 256 + tid;
    const float qx = sx[p], qy = sy[p];
    float bd[KK]; int bi[KK];
#pragma unroll
    for (int t = 0; t < KK; ++t) { bd[t] = 3.4e38f; bi[t] = t; }
    for (int j = 0; j < PP; ++j) {
        const float dx = qx - sx[j], dy = qy - sy[j];
        const float d = sm[j] ? (dx * dx + dy * dy) : 3.0e38f;
        if (d < bd[KK - 1]) {
            int pos = KK - 1;
            while (pos > 0 && d < bd[pos - 1]) {
                bd[pos] = bd[pos - 1]; bi[pos] = bi[pos - 1]; --pos;
            }
            bd[pos] = d; bi[pos] = j;
        }
    }
    int* op = idx + ((size_t)b * PP + p) * KK;
#pragma unroll
    for (int t = 0; t < KK; ++t) op[t] = bi[t];
}

// ---------------------------------------------------------------------------
// Fused: neighbor max-pool + f16 packing of cat512 (enriched|mp) and
// comb352 (enriched | msg(later) | skip | zero-pad). grid B*P, block 256.
// ---------------------------------------------------------------------------
__global__ void ttf_pack(const float* __restrict__ enr,
                         const float* __restrict__ feat,
                         const float* __restrict__ lor,
                         const unsigned char* __restrict__ mask,
                         const int* __restrict__ idx,
                         const float* __restrict__ skg,
                         const float* __restrict__ skb,
                         _Float16* __restrict__ cat512,
                         _Float16* __restrict__ comb)
{
    const int bp = blockIdx.x;
    const int b = bp >> 11;
    const int p = bp & (PP - 1);
    const int c = threadIdx.x;
    const size_t rowc = (size_t)bp * 512;
    const size_t rowm = (size_t)bp * COMBP;

    const float e = enr[((size_t)b * CC + c) * PP + p];
    cat512[rowc + c] = (_Float16)e;
    comb[rowm + c]   = (_Float16)e;

    const int* ip = idx + (size_t)bp * KK;
    float m = -3.4e38f;
#pragma unroll
    for (int k = 0; k < KK; ++k) {
        const int j = ip[k];
        if (mask[b * PP + j]) {
            m = fmaxf(m, enr[((size_t)b * CC + c) * PP + j]);
        }
    }
    if (m < -3.0e38f) m = 0.0f;
    cat512[rowc + CC + c] = (_Float16)m;

    const float mf = mask[b * PP + p] ? 1.0f : 0.0f;
    const float rs = rsqrtf(1.0f + EPSV);
    if (c == 0) {
        const float f = feat[((size_t)b * 8 + 6) * PP + p];
        comb[rowm + 320] = (_Float16)((f * skg[0] * rs + skb[0]) * mf);
    }
    if (c == 1) {
        const float lx = lor[((size_t)b * 4 + 0) * PP + p];
        const float ly = lor[((size_t)b * 4 + 1) * PP + p];
        const float pt = sqrtf(lx * lx + ly * ly + 1e-8f);
        comb[rowm + 321] = (_Float16)((pt * skg[1] * rs + skb[1]) * mf);
    }
    if (c < COMBP - 322) comb[rowm + 322 + c] = (_Float16)0.0f;
}

// msg epilogue: gelu(bn(msg_raw)) * mask -> comb f16 channels [256,320)
__global__ void ttf_msg_post(const float* __restrict__ msg,
                             const float* __restrict__ g,
                             const float* __restrict__ bb,
                             const unsigned char* __restrict__ mask,
                             _Float16* __restrict__ comb)
{
    const int i = blockIdx.x * 256 + threadIdx.x;  // B*64*P
    const int b = i >> 17;
    const int rem = i & 131071;
    const int o = rem >> 11;
    const int p = rem & (PP - 1);
    const float rs = rsqrtf(1.0f + EPSV);
    float v = msg[i] * g[o] * rs + bb[o];
    v = gelu_f(v);
    v *= mask[b * PP + p] ? 1.0f : 0.0f;
    comb[((size_t)b * PP + p) * COMBP + CC + o] = (_Float16)v;
}

// per-track head: sum_o h2_w[o]*gelu(bn(h_raw)) + h2_b -> d_out (B,P)
__global__ void ttf_pertrack(const float* __restrict__ h,
                             const float* __restrict__ g,
                             const float* __restrict__ bb,
                             const float* __restrict__ w2,
                             const float* __restrict__ b2,
                             float* __restrict__ out)
{
    const int b = blockIdx.y;
    const int p = blockIdx.x * 256 + threadIdx.x;
    const float rs = rsqrtf(1.0f + EPSV);
    float s = 0.0f;
    for (int o = 0; o < DD; ++o) {
        float v = h[((size_t)b * DD + o) * PP + p];
        v = v * g[o] * rs + bb[o];
        s += w2[o] * gelu_f(v);
    }
    out[(size_t)b * PP + p] = s + b2[0];
}

// top-KC selection (iterative argmax), one block per batch
__global__ void ttf_topk(const float* __restrict__ pt,
                         const unsigned char* __restrict__ mask,
                         int* __restrict__ topIdx)
{
    __shared__ float sv[PP];
    __shared__ float rv[256];
    __shared__ int   ri[256];
    const int b = blockIdx.x;
    const int tid = threadIdx.x;
    for (int j = tid; j < PP; j += 256)
        sv[j] = mask[b * PP + j] ? pt[(size_t)b * PP + j] : -1e9f;
    __syncthreads();
    for (int it = 0; it < KCC; ++it) {
        float bv = -3.4e38f; int bidx = 0;
        for (int j = tid * 8; j < tid * 8 + 8; ++j) {
            if (sv[j] > bv) { bv = sv[j]; bidx = j; }
        }
        rv[tid] = bv; ri[tid] = bidx;
        __syncthreads();
        for (int s = 128; s > 0; s >>= 1) {
            if (tid < s) {
                if (rv[tid + s] > rv[tid] ||
                    (rv[tid + s] == rv[tid] && ri[tid + s] < ri[tid])) {
                    rv[tid] = rv[tid + s]; ri[tid] = ri[tid + s];
                }
            }
            __syncthreads();
        }
        if (tid == 0) { topIdx[b * KCC + it] = ri[0]; sv[ri[0]] = -3.4e38f; }
        __syncthreads();
    }
}

// gather candidate rows (f16, 352 channels), block 352
__global__ void ttf_gather(const _Float16* __restrict__ comb,
                           const int* __restrict__ topIdx,
                           _Float16* __restrict__ cand)
{
    const int bt = blockIdx.x;
    const int b = bt >> 8;
    const int c = threadIdx.x;
    const int p = topIdx[bt];
    cand[(size_t)bt * COMBP + c] = comb[((size_t)b * PP + p) * COMBP + c];
}

// LayerNorm over D=128, one block per token, block 128
__global__ void ttf_ln(const float* __restrict__ src,
                       const float* __restrict__ g,
                       const float* __restrict__ b,
                       float* __restrict__ dst,
                       _Float16* __restrict__ dsth)
{
    __shared__ float red[DD];
    const int t = blockIdx.x;
    const int i = threadIdx.x;
    const float v = src[(size_t)t * DD + i];
    red[i] = v; __syncthreads();
    for (int s = 64; s > 0; s >>= 1) { if (i < s) red[i] += red[i + s]; __syncthreads(); }
    const float mean = red[0] / (float)DD;
    __syncthreads();
    const float d = v - mean;
    red[i] = d * d; __syncthreads();
    for (int s = 64; s > 0; s >>= 1) { if (i < s) red[i] += red[i + s]; __syncthreads(); }
    const float var = red[0] / (float)DD;
    const float y = d * rsqrtf(var + EPSV) * g[i] + b[i];
    if (dst)  dst[(size_t)t * DD + i]  = y;
    if (dsth) dsth[(size_t)t * DD + i] = (_Float16)y;
}

// split qkv_raw -> q_h (B,H,KC,32), k_h (B,H,KC,32), v_t (B,H,32,KC); block 384
__global__ void ttf_qkv_split(const float* __restrict__ qkv,
                              _Float16* __restrict__ q,
                              _Float16* __restrict__ k,
                              _Float16* __restrict__ v)
{
    const int t = blockIdx.x;          // b*KC + tok
    const int b = t >> 8;
    const int tok = t & 255;
    const int c = threadIdx.x;
    const float val = qkv[(size_t)t * 384 + c];
    const int which = c / DD;
    const int cc = c % DD;
    const int h = cc / 32;
    const int d = cc % 32;
    const size_t hb = (size_t)(b * HH + h);
    if (which == 0)      q[(hb * KCC + tok) * 32 + d] = (_Float16)val;
    else if (which == 1) k[(hb * KCC + tok) * 32 + d] = (_Float16)val;
    else                 v[(hb * 32 + d) * KCC + tok] = (_Float16)val;
}

// softmax over rows of 256, output f16; one block per row
__global__ void ttf_softmax(const float* __restrict__ s, _Float16* __restrict__ p)
{
    __shared__ float red[256];
    const size_t row = blockIdx.x;
    const int i = threadIdx.x;
    const float v = s[row * 256 + i];
    red[i] = v; __syncthreads();
    for (int st = 128; st > 0; st >>= 1) { if (i < st) red[i] = fmaxf(red[i], red[i + st]); __syncthreads(); }
    const float mx = red[0];
    __syncthreads();
    const float e = expf(v - mx);
    red[i] = e; __syncthreads();
    for (int st = 128; st > 0; st >>= 1) { if (i < st) red[i] += red[i + st]; __syncthreads(); }
    p[row * 256 + i] = (_Float16)(e / red[0]);
}

// generic f32 -> f16 convert
__global__ void ttf_cvt(const float* __restrict__ src, _Float16* __restrict__ dst, int n)
{
    const int i = blockIdx.x * 256 + threadIdx.x;
    if (i < n) dst[i] = (_Float16)src[i];
}

// weight convert with K-padding: dst (rows x kpad), src (rows x kin)
__global__ void ttf_cvt_pad(const float* __restrict__ src, _Float16* __restrict__ dst,
                            int rows, int kin, int kpad)
{
    const int i = blockIdx.x * 256 + threadIdx.x;
    if (i >= rows * kpad) return;
    const int r = i / kpad, k = i % kpad;
    dst[i] = (k < kin) ? (_Float16)src[(size_t)r * kin + k] : (_Float16)0.0f;
}

// refined = gelu-MLP head output dot s2 + b2; scatter-add into d_out
__global__ void ttf_scatter(const float* __restrict__ sc,
                            const float* __restrict__ w2,
                            const float* __restrict__ b2,
                            const int* __restrict__ topIdx,
                            float* __restrict__ out)
{
    const int b = blockIdx.x;
    const int t = threadIdx.x;
    float s = 0.0f;
    for (int o = 0; o < 64; ++o)
        s += sc[((size_t)(b * KCC + t)) * 64 + o] * w2[o];
    s += b2[0];
    out[(size_t)b * PP + topIdx[b * KCC + t]] += s;
}

// ---------------------------------------------------------------------------
// Host side
// ---------------------------------------------------------------------------
extern "C" void kernel_launch(void* const* d_in, const int* in_sizes, int n_in,
                              void* d_out, int out_size, void* d_ws, size_t ws_size,
                              hipStream_t stream)
{
    (void)in_sizes; (void)n_in; (void)out_size; (void)ws_size;

    const float* enriched = (const float*)d_in[0];
    const float* features = (const float*)d_in[1];
    const float* lorentz  = (const float*)d_in[2];
    const float* points   = (const float*)d_in[3];
    const unsigned char* mask = (const unsigned char*)d_in[4];
    const float* msg_w = (const float*)d_in[5];
    const float* msg_g = (const float*)d_in[6];
    const float* msg_b = (const float*)d_in[7];
    const float* skip_g = (const float*)d_in[8];
    const float* skip_b = (const float*)d_in[9];
    const float* h1_w = (const float*)d_in[10];
    const float* h1_g = (const float*)d_in[11];
    const float* h1_b = (const float*)d_in[12];
    const float* h2_w = (const float*)d_in[13];
    const float* h2_b = (const float*)d_in[14];
    const float* proj_w = (const float*)d_in[15];
    const float* proj_b = (const float*)d_in[16];
    const float* ln0_g = (const float*)d_in[17];
    const float* ln0_b = (const float*)d_in[18];
    const float* ln1_g = (const float*)d_in[19];
    const float* ln1_b = (const float*)d_in[20];
    const float* qkv_w = (const float*)d_in[21];
    const float* qkv_b = (const float*)d_in[22];
    const float* ao_w = (const float*)d_in[23];
    const float* ao_b = (const float*)d_in[24];
    const float* ln2_g = (const float*)d_in[25];
    const float* ln2_b = (const float*)d_in[26];
    const float* ff1_w = (const float*)d_in[27];
    const float* ff1_b = (const float*)d_in[28];
    const float* ff2_w = (const float*)d_in[29];
    const float* ff2_b = (const float*)d_in[30];
    const float* s1_w = (const float*)d_in[31];
    const float* s1_b = (const float*)d_in[32];
    const float* s2_w = (const float*)d_in[33];
    const float* s2_b = (const float*)d_in[34];

    float* out = (float*)d_out;

    // ---- workspace carve-up ----
    char* base = (char*)d_ws;
    size_t off = 0;
    auto alloc = [&](size_t bytes) -> void* {
        off = (off + 255) & ~(size_t)255;
        void* p = base + off;
        off += bytes;
        return p;
    };

    int*       idx_knn = (int*)      alloc((size_t)BB * PP * KK * 4);
    _Float16*  cat512  = (_Float16*) alloc((size_t)BB * PP * 512 * 2);
    _Float16*  comb    = (_Float16*) alloc((size_t)BB * PP * COMBP * 2);
    float*     msg_raw = (float*)    alloc((size_t)BB * MSGC * PP * 4);
    float*     h_raw   = (float*)    alloc((size_t)BB * DD * PP * 4);
    _Float16*  wh_msg  = (_Float16*) alloc((size_t)MSGC * 512 * 2);
    _Float16*  wh_h1   = (_Float16*) alloc((size_t)DD * COMBP * 2);
    _Float16*  wh_proj = (_Float16*) alloc((size_t)DD * COMBP * 2);
    _Float16*  wh_qkv  = (_Float16*) alloc((size_t)NLL * 384 * DD * 2);
    _Float16*  wh_ao   = (_Float16*) alloc((size_t)NLL * DD * DD * 2);
    _Float16*  wh_ff1  = (_Float16*) alloc((size_t)NLL * 512 * DD * 2);
    _Float16*  wh_ff2  = (_Float16*) alloc((size_t)NLL * DD * 512 * 2);
    _Float16*  wh_s1   = (_Float16*) alloc((size_t)64 * DD * 2);
    int*       top_idx = (int*)      alloc((size_t)BB * KCC * 4);
    _Float16*  cand    = (_Float16*) alloc((size_t)BB * KCC * COMBP * 2);
    float*     x       = (float*)    alloc((size_t)BB * KCC * DD * 4);
    _Float16*  y_h     = (_Float16*) alloc((size_t)BB * KCC * DD * 2);
    float*     qkv_raw = (float*)    alloc((size_t)BB * KCC * 384 * 4);
    _Float16*  q_h     = (_Float16*) alloc((size_t)BB * HH * KCC * 32 * 2);
    _Float16*  k_h     = (_Float16*) alloc((size_t)BB * HH * KCC * 32 * 2);
    _Float16*  v_t     = (_Float16*) alloc((size_t)BB * HH * 32 * KCC * 2);
    float*     scores  = (float*)    alloc((size_t)BB * HH * KCC * KCC * 4);
    _Float16*  probs_h = (_Float16*) alloc((size_t)BB * HH * KCC * KCC * 2);
    float*     o_raw   = (float*)    alloc((size_t)BB * KCC * DD * 4);
    _Float16*  o_h     = (_Float16*) alloc((size_t)BB * KCC * DD * 2);
    float*     ff_raw  = (float*)    alloc((size_t)BB * KCC * 512 * 4);
    _Float16*  ff_h    = (_Float16*) alloc((size_t)BB * KCC * 512 * 2);
    _Float16*  x_h     = (_Float16*) alloc((size_t)BB * KCC * DD * 2);
    float*     sc      = (float*)    alloc((size_t)BB * KCC * 64 * 4);

    auto cvtpad = [&](const float* src, _Float16* dst, int rows, int kin, int kpad) {
        int n = rows * kpad;
        ttf_cvt_pad<<<dim3((n + 255) / 256), dim3(256), 0, stream>>>(src, dst, rows, kin, kpad);
    };
    auto gemm = [&](const _Float16* A, long long sA, int lda,
                    const _Float16* Bm, long long sB, int ldb,
                    float* Cm, long long sC1, long long sC2, int zdiv, int ldc,
                    int M, int N, int Kp, int Z,
                    const float* bias, float scale, int flags) {
        dim3 g(N / 16, (M + 63) / 64, Z);
        ttf_gemm<<<g, dim3(128), 0, stream>>>(A, sA, lda, Bm, sB, ldb,
                                              Cm, sC1, sC2, zdiv, ldc,
                                              M, N, Kp, bias, scale, flags);
    };

    // ---- weight conversions (f32 -> f16, with K padding where needed) ----
    cvtpad(msg_w, wh_msg, MSGC, 512, 512);
    cvtpad(h1_w, wh_h1, DD, 322, COMBP);
    cvtpad(proj_w, wh_proj, DD, 322, COMBP);
    cvtpad(qkv_w, wh_qkv, NLL * 384, DD, DD);
    cvtpad(ao_w, wh_ao, NLL * DD, DD, DD);
    cvtpad(ff1_w, wh_ff1, NLL * 512, DD, DD);
    cvtpad(ff2_w, wh_ff2, NLL * DD, 512, 512);
    cvtpad(s1_w, wh_s1, 64, DD, DD);

    // ---- M1: kNN + gather/max-pool + pack activations ----
    ttf_knn<<<dim3(BB, PP / 256), dim3(256), 0, stream>>>(points, mask, idx_knn);
    ttf_pack<<<dim3(BB * PP), dim3(256), 0, stream>>>(
        enriched, features, lorentz, mask, idx_knn, skip_g, skip_b, cat512, comb);

    // msg GEMM: (64 x 512) x (P x 512)^T per batch
    gemm(wh_msg, 0, 512, cat512, (long long)PP * 512, 512,
         msg_raw, (long long)MSGC * PP, 0, 1, PP,
         MSGC, PP, 512, BB, nullptr, 1.0f, 0);
    ttf_msg_post<<<dim3((BB * MSGC * PP) / 256), dim3(256), 0, stream>>>(
        msg_raw, msg_g, msg_b, mask, comb);

    // h1 GEMM: (128 x 352) x (P x 352)^T per batch
    gemm(wh_h1, 0, COMBP, comb, (long long)PP * COMBP, COMBP,
         h_raw, (long long)DD * PP, 0, 1, PP,
         DD, PP, COMBP, BB, nullptr, 1.0f, 0);
    ttf_pertrack<<<dim3(PP / 256, BB), dim3(256), 0, stream>>>(
        h_raw, h1_g, h1_b, h2_w, h2_b, out);

    // ---- M2: top-KC selection + gather + projection + LN0 ----
    ttf_topk<<<dim3(BB), dim3(256), 0, stream>>>(out, mask, top_idx);
    ttf_gather<<<dim3(BB * KCC), dim3(COMBP), 0, stream>>>(comb, top_idx, cand);
    gemm(cand, (long long)KCC * COMBP, COMBP, wh_proj, 0, COMBP,
         x, (long long)KCC * DD, 0, 1, DD,
         KCC, DD, COMBP, BB, proj_b, 1.0f, 0);
    ttf_ln<<<dim3(BB * KCC), dim3(DD), 0, stream>>>(x, ln0_g, ln0_b, x, (_Float16*)nullptr);

    // ---- transformer layers ----
    const float inv_sqrt_dh = 0.17677669529663687f;  // 1/sqrt(32)
    for (int l = 0; l < NLL; ++l) {
        // y = LN1(x) -> f16
        ttf_ln<<<dim3(BB * KCC), dim3(DD), 0, stream>>>(
            x, ln1_g + l * DD, ln1_b + l * DD, (float*)nullptr, y_h);
        // qkv = y @ qkv_w^T + b
        gemm(y_h, (long long)KCC * DD, DD, wh_qkv + (size_t)l * 384 * DD, 0, DD,
             qkv_raw, (long long)KCC * 384, 0, 1, 384,
             KCC, 384, DD, BB, qkv_b + l * 384, 1.0f, 0);
        ttf_qkv_split<<<dim3(BB * KCC), dim3(384), 0, stream>>>(qkv_raw, q_h, k_h, v_t);
        // scores = Q K^T / sqrt(dh), per (b,h)
        gemm(q_h, (long long)KCC * 32, 32, k_h, (long long)KCC * 32, 32,
             scores, (long long)KCC * KCC, 0, 1, KCC,
             KCC, KCC, 32, BB * HH, nullptr, inv_sqrt_dh, 0);
        ttf_softmax<<<dim3(BB * HH * KCC), dim3(256), 0, stream>>>(scores, probs_h);
        // O = P V, written into o_raw at head offset
        gemm(probs_h, (long long)KCC * KCC, KCC, v_t, (long long)32 * KCC, KCC,
             o_raw, (long long)KCC * DD, 32, HH, DD,
             KCC, 32, KCC, BB * HH, nullptr, 1.0f, 0);
        ttf_cvt<<<dim3((BB * KCC * DD) / 256), dim3(256), 0, stream>>>(o_raw, o_h, BB * KCC * DD);
        // x += O @ ao_w^T + ao_b
        gemm(o_h, (long long)KCC * DD, DD, wh_ao + (size_t)l * DD * DD, 0, DD,
             x, (long long)KCC * DD, 0, 1, DD,
             KCC, DD, DD, BB, ao_b + l * DD, 1.0f, GF_ACC);
        // y = LN2(x) -> f16
        ttf_ln<<<dim3(BB * KCC), dim3(DD), 0, stream>>>(
            x, ln2_g + l * DD, ln2_b + l * DD, (float*)nullptr, y_h);
        // ff = gelu(y @ ff1_w^T + b)
        gemm(y_h, (long long)KCC * DD, DD, wh_ff1 + (size_t)l * 512 * DD, 0, DD,
             ff_raw, (long long)KCC * 512, 0, 1, 512,
             KCC, 512, DD, BB, ff1_b + l * 512, 1.0f, GF_GELU);
        ttf_cvt<<<dim3((BB * KCC * 512) / 256), dim3(256), 0, stream>>>(ff_raw, ff_h, BB * KCC * 512);
        // x += ff @ ff2_w^T + b
        gemm(ff_h, (long long)KCC * 512, 512, wh_ff2 + (size_t)l * DD * 512, 0, 512,
             x, (long long)KCC * DD, 0, 1, DD,
             KCC, DD, 512, BB, ff2_b + l * DD, 1.0f, GF_ACC);
    }

    // ---- refinement head + scatter-add ----
    ttf_cvt<<<dim3((BB * KCC * DD) / 256), dim3(256), 0, stream>>>(x, x_h, BB * KCC * DD);
    gemm(x_h, (long long)KCC * DD, DD, wh_s1, 0, DD,
         sc, (long long)KCC * 64, 0, 1, 64,
         KCC, 64, DD, BB, s1_b, 1.0f, GF_GELU);
    ttf_scatter<<<dim3(BB), dim3(KCC), 0, stream>>>(sc, s2_w, s2_b, top_idx, out);
}
